// Transformer_3255585210708
// MI455X (gfx1250) — compile-verified, matrix operand order checked by reading, not run on hardware
//
#include <hip/hip_runtime.h>

// ---------------------------------------------------------------------------
// GPT-style 6-layer transformer forward for gfx1250 (MI455X).
// All GEMMs via v_wmma_f32_16x16x32_f16 with f16 operands end-to-end on the
// matmul path (f16 weights + f16 activations, fp32 accumulate). K-loop is
// ping-pong software-pipelined (unroll-by-2, loads land directly in WMMA
// operand registers — no rotation movs). Flash-style attention, fp32
// residual stream / LayerNorm / softmax.
// ---------------------------------------------------------------------------

typedef __attribute__((ext_vector_type(16))) _Float16 v16h;
typedef __attribute__((ext_vector_type(8)))  _Float16 v8h;
typedef __attribute__((ext_vector_type(8)))  float    v8f;

union Frag16 { v16h v; v8h h[2]; };

constexpr int L  = 6;
constexpr int D  = 768;
constexpr int H  = 12;
constexpr int HD = 64;
constexpr int B  = 8;
constexpr int S  = 1024;
constexpr int M  = B * S;        // 8192 token rows
constexpr int D3 = 3 * D;        // 2304
constexpr int D4 = 4 * D;        // 3072

__device__ __forceinline__ float red_max16(float v) {
  #pragma unroll
  for (int m = 1; m < 16; m <<= 1) v = fmaxf(v, __shfl_xor(v, m, 32));
  return v;
}
__device__ __forceinline__ float red_sum16(float v) {
  #pragma unroll
  for (int m = 1; m < 16; m <<= 1) v += __shfl_xor(v, m, 32);
  return v;
}
__device__ __forceinline__ float gelu_f(float x) {
  const float k = 0.7978845608028654f;  // sqrt(2/pi)
  float t = tanhf(k * (x + 0.044715f * x * x * x));
  return 0.5f * x * (1.0f + t);
}

// --------------------------- weight fp32 -> f16 ----------------------------
__global__ void f32_to_f16_kernel(const float* __restrict__ src,
                                  _Float16* __restrict__ dst, size_t n) {
  size_t i = (size_t)blockIdx.x * blockDim.x + threadIdx.x;
  size_t stride = (size_t)gridDim.x * blockDim.x;
  for (; i < n; i += stride) dst[i] = (_Float16)src[i];
}

// ------------------------------- embedding ---------------------------------
__global__ void embed_kernel(const int* __restrict__ tokens,
                             const float* __restrict__ tok_emb,
                             const float* __restrict__ pos_emb,
                             float* __restrict__ x, _Float16* __restrict__ xh) {
  __shared__ int poss[S];
  const int b = blockIdx.x;
  if (threadIdx.x == 0) {
    int run = 0;
    for (int s = 0; s < S; ++s) {
      int t = tokens[b * S + s];
      if (t != 0) { ++run; poss[s] = run; } else { poss[s] = 0; }
    }
  }
  __syncthreads();
  for (int idx = threadIdx.x; idx < S * D; idx += blockDim.x) {
    int s = idx / D, d = idx - s * D;
    int t = tokens[b * S + s];
    float v = tok_emb[(size_t)t * D + d] + pos_emb[(size_t)poss[s] * D + d];
    size_t o = ((size_t)b * S + s) * D + d;
    x[o] = v;
    xh[o] = (_Float16)v;
  }
}

// ------------------------ WMMA GEMM: Y = A @ W^T + b -----------------------
// A: [M,K] f16   W: [N,K] f16   out fp32 (Yf) or f16 (Yh).  act: 1 = gelu.
// block: 256 thr = 8 waves (2 M-strips x 4 N-strips); wave tile 32x64
// (2x4 C tiles, 8 WMMAs / 32-chunk); block tile 64x256.
// K-loop: ping-pong unroll-by-2, branch-free interior (K % 64 == 0).
__global__ __launch_bounds__(256)
void gemm_wmma_kernel(const _Float16* __restrict__ A,
                      const _Float16* __restrict__ W,
                      const float* __restrict__ bias,
                      float* __restrict__ Yf, _Float16* __restrict__ Yh,
                      int N, int K, int act, int out16) {
  const int lane = threadIdx.x & 31;
  const int wave = threadIdx.x >> 5;
  const int ln = lane & 15, hi = lane >> 4;
  const int wm = wave & 1, wn = wave >> 1;
  const int rowBase = blockIdx.y * 64 + wm * 32;
  const int colBase = blockIdx.x * 256 + wn * 64;

  const _Float16* Arow[2];
  #pragma unroll
  for (int i = 0; i < 2; ++i)
    Arow[i] = A + (size_t)(rowBase + i * 16 + ln) * K + hi * 8;
  const _Float16* Brow[4];
  #pragma unroll
  for (int t = 0; t < 4; ++t)
    Brow[t] = W + (size_t)(colBase + t * 16 + ln) * K + hi * 8;

  #pragma unroll
  for (int t = 0; t < 4; ++t) __builtin_prefetch(Brow[t], 0, 1);

  v8f acc[2][4];
  #pragma unroll
  for (int i = 0; i < 2; ++i)
    #pragma unroll
    for (int t = 0; t < 4; ++t)
      #pragma unroll
      for (int r = 0; r < 8; ++r) acc[i][t][r] = 0.f;

  Frag16 fa0[2], fb0[4], fa1[2], fb1[4];

  auto loadAB = [&](Frag16* fa, Frag16* fb, int k) {
    #pragma unroll
    for (int i = 0; i < 2; ++i) {
      fa[i].h[0] = *(const v8h*)(Arow[i] + k);
      fa[i].h[1] = *(const v8h*)(Arow[i] + k + 16);
    }
    #pragma unroll
    for (int t = 0; t < 4; ++t) {
      fb[t].h[0] = *(const v8h*)(Brow[t] + k);
      fb[t].h[1] = *(const v8h*)(Brow[t] + k + 16);
    }
  };
  auto mmastep = [&](Frag16* fa, Frag16* fb) {
    #pragma unroll
    for (int t = 0; t < 4; ++t)
      #pragma unroll
      for (int i = 0; i < 2; ++i)
        acc[i][t] = __builtin_amdgcn_wmma_f32_16x16x32_f16(
            false, fa[i].v, false, fb[t].v, (short)0, acc[i][t], false, false);
  };

  loadAB(fa0, fb0, 0);
  int k0 = 0;
  // interior: both prefetches in-range (K % 64 == 0)
  for (; k0 + 64 < K; k0 += 64) {
    loadAB(fa1, fb1, k0 + 32);
    mmastep(fa0, fb0);
    loadAB(fa0, fb0, k0 + 64);
    mmastep(fa1, fb1);
  }
  // tail: k0 == K - 64
  loadAB(fa1, fb1, k0 + 32);
  mmastep(fa0, fb0);
  mmastep(fa1, fb1);

  #pragma unroll
  for (int i = 0; i < 2; ++i)
    #pragma unroll
    for (int t = 0; t < 4; ++t) {
      int col = colBase + t * 16 + ln;
      float bv = bias[col];
      #pragma unroll
      for (int r = 0; r < 8; ++r) {
        int row = rowBase + i * 16 + r + 8 * hi;
        float v = acc[i][t][r] + bv;
        if (act == 1) v = gelu_f(v);
        if (out16) Yh[(size_t)row * N + col] = (_Float16)v;
        else       Yf[(size_t)row * N + col] = v;
      }
    }
}

// ------------------------- flash attention (1 wave) ------------------------
// qkv: [B*S, 3D] f16 (q|k|v).  aout: [B*S, D] f16.
// grid: (S/16, H, B); block: 32 (one wave owns a 16-query x HD=64 tile).
__global__ __launch_bounds__(32)
void attn_kernel(const _Float16* __restrict__ qkv,
                 const int* __restrict__ tokens,
                 _Float16* __restrict__ aout) {
  const int qt = blockIdx.x, h = blockIdx.y, b = blockIdx.z;
  const int q0 = qt * 16;
  const int lane = threadIdx.x & 31, ln = lane & 15, hi = lane >> 4;
  __shared__ _Float16 plds[16][34];

  // Q fragments (K-dim = HD = 64 -> two 32-chunks), pre-scaled by 1/sqrt(HD)
  v16h aq[2];
  {
    const _Float16* Qrow =
        qkv + (size_t)(b * S + q0 + ln) * D3 + h * HD + hi * 8;
    #pragma unroll
    for (int c = 0; c < 2; ++c) {
      v8h l0 = *(const v8h*)(Qrow + c * 32);
      v8h l1 = *(const v8h*)(Qrow + c * 32 + 16);
      #pragma unroll
      for (int j = 0; j < 8; ++j) {
        aq[c][j]     = l0[j] * (_Float16)0.125f;
        aq[c][8 + j] = l1[j] * (_Float16)0.125f;
      }
    }
  }

  float rowmax[8], rowsum[8];
  v8f acc[4];
  #pragma unroll
  for (int r = 0; r < 8; ++r) { rowmax[r] = -1e30f; rowsum[r] = 0.f; }
  #pragma unroll
  for (int u = 0; u < 4; ++u)
    #pragma unroll
    for (int r = 0; r < 8; ++r) acc[u][r] = 0.f;

  const int nch = (q0 + 16 + 31) / 32;  // causal: only keys <= q0+15 needed
  for (int kc = 0; kc < nch; ++kc) {
    const int kb = kc * 32;

    // scores tile: 16 q x 32 k  (two 16x16 C tiles)
    v8f sc[2];
    #pragma unroll
    for (int t = 0; t < 2; ++t)
      #pragma unroll
      for (int r = 0; r < 8; ++r) sc[t][r] = 0.f;

    #pragma unroll
    for (int t = 0; t < 2; ++t) {
      const _Float16* Krow =
          qkv + (size_t)(b * S + kb + t * 16 + ln) * D3 + D + h * HD + hi * 8;
      #pragma unroll
      for (int c = 0; c < 2; ++c) {
        Frag16 bk;
        bk.h[0] = *(const v8h*)(Krow + c * 32);
        bk.h[1] = *(const v8h*)(Krow + c * 32 + 16);
        sc[t] = __builtin_amdgcn_wmma_f32_16x16x32_f16(
            false, aq[c], false, bk.v, (short)0, sc[t], false, false);
      }
    }

    // masking + online softmax
    const int kp0 = kb + ln, kp1 = kb + 16 + ln;
    const int pad0 = (tokens[b * S + kp0] == 0);
    const int pad1 = (tokens[b * S + kp1] == 0);
    float nm[8], sscale[8];
    #pragma unroll
    for (int r = 0; r < 8; ++r) {
      int qi = q0 + r + 8 * hi;
      float s0 = (kp0 > qi || pad0) ? -1e30f : sc[0][r];
      float s1 = (kp1 > qi || pad1) ? -1e30f : sc[1][r];
      sc[0][r] = s0; sc[1][r] = s1;
      float m = red_max16(fmaxf(s0, s1));
      nm[r] = fmaxf(rowmax[r], m);
      sscale[r] = __expf(rowmax[r] - nm[r]);
      rowmax[r] = nm[r];
    }
    #pragma unroll
    for (int r = 0; r < 8; ++r) {
      float p0 = (sc[0][r] <= -1e29f) ? 0.f : __expf(sc[0][r] - nm[r]);
      float p1 = (sc[1][r] <= -1e29f) ? 0.f : __expf(sc[1][r] - nm[r]);
      float ps = red_sum16(p0 + p1);
      rowsum[r] = rowsum[r] * sscale[r] + ps;
      #pragma unroll
      for (int u = 0; u < 4; ++u) acc[u][r] *= sscale[r];
      plds[r + 8 * hi][ln]      = (_Float16)p0;
      plds[r + 8 * hi][16 + ln] = (_Float16)p1;
    }
    __syncthreads();

    // P (16x32) re-laid as A-fragment via LDS
    v16h ap;
    #pragma unroll
    for (int j = 0; j < 16; ++j) {
      int kk = (j < 8) ? (hi * 8 + j) : (16 + hi * 8 + (j - 8));
      ap[j] = plds[ln][kk];
    }
    __syncthreads();

    // O += P @ V  (V chunk 32 x 64 -> 4 B tiles)
    #pragma unroll
    for (int u = 0; u < 4; ++u) {
      v16h bv;
      #pragma unroll
      for (int j = 0; j < 16; ++j) {
        int kk = (j < 8) ? (hi * 8 + j) : (16 + hi * 8 + (j - 8));
        bv[j] = qkv[(size_t)(b * S + kb + kk) * D3 + 2 * D +
                    h * HD + u * 16 + ln];
      }
      acc[u] = __builtin_amdgcn_wmma_f32_16x16x32_f16(
          false, ap, false, bv, (short)0, acc[u], false, false);
    }
  }

  // normalize; fully-masked rows (rowsum==0) -> 0 per reference
  #pragma unroll
  for (int r = 0; r < 8; ++r) {
    int qi = q0 + r + 8 * hi;
    float inv = (rowsum[r] > 0.f) ? (1.f / rowsum[r]) : 0.f;
    #pragma unroll
    for (int u = 0; u < 4; ++u)
      aout[(size_t)(b * S + qi) * D + h * HD + u * 16 + ln] =
          (_Float16)(acc[u][r] * inv);
  }
}

// ------------- x = LayerNorm(x + a) * g + b  (fp32 + f16 mirror) -----------
__global__ __launch_bounds__(256)
void add_ln_kernel(const float* __restrict__ xin, const float* __restrict__ add,
                   const float* __restrict__ g, const float* __restrict__ bta,
                   float* __restrict__ xout, _Float16* __restrict__ xhout) {
  __shared__ float red[256];
  const int row = blockIdx.x;
  const int tid = threadIdx.x;
  float v[3];
  #pragma unroll
  for (int i = 0; i < 3; ++i) {
    int d = tid + i * 256;
    v[i] = xin[(size_t)row * D + d] + add[(size_t)row * D + d];
  }
  float s = v[0] + v[1] + v[2];
  red[tid] = s; __syncthreads();
  for (int off = 128; off > 0; off >>= 1) {
    if (tid < off) red[tid] += red[tid + off];
    __syncthreads();
  }
  float mu = red[0] / (float)D;
  __syncthreads();
  float q = 0.f;
  #pragma unroll
  for (int i = 0; i < 3; ++i) { float d = v[i] - mu; q += d * d; }
  red[tid] = q; __syncthreads();
  for (int off = 128; off > 0; off >>= 1) {
    if (tid < off) red[tid] += red[tid + off];
    __syncthreads();
  }
  float var = red[0] / (float)D;
  float rinv = rsqrtf(var + 1e-5f);
  #pragma unroll
  for (int i = 0; i < 3; ++i) {
    int d = tid + i * 256;
    float o = (v[i] - mu) * rinv * g[d] + bta[d];
    xout[(size_t)row * D + d] = o;
    xhout[(size_t)row * D + d] = (_Float16)o;
  }
}

// ------------------------------- outputs -----------------------------------
__global__ void copy_kernel(const float* __restrict__ src, float* __restrict__ dst,
                            size_t n) {
  size_t i = (size_t)blockIdx.x * blockDim.x + threadIdx.x;
  size_t stride = (size_t)gridDim.x * blockDim.x;
  for (; i < n; i += stride) dst[i] = src[i];
}
__global__ void pad_out_kernel(const int* __restrict__ tokens,
                               float* __restrict__ dst, int n) {
  int i = blockIdx.x * blockDim.x + threadIdx.x;
  if (i < n) dst[i] = (tokens[i] == 0) ? 1.0f : 0.0f;
}

// ---------------------------------------------------------------------------
extern "C" void kernel_launch(void* const* d_in, const int* in_sizes, int n_in,
                              void* d_out, int out_size, void* d_ws, size_t ws_size,
                              hipStream_t stream) {
  (void)in_sizes; (void)n_in; (void)out_size; (void)ws_size;

  const int*   tokens  = (const int*)d_in[0];
  const float* tok_emb = (const float*)d_in[1];
  const float* pos_emb = (const float*)d_in[2];
  const float* qkv_w   = (const float*)d_in[3];
  const float* qkv_b   = (const float*)d_in[4];
  const float* out_w   = (const float*)d_in[5];
  const float* out_b   = (const float*)d_in[6];
  const float* ln1_g   = (const float*)d_in[7];
  const float* ln1_b   = (const float*)d_in[8];
  const float* ff1_w   = (const float*)d_in[9];
  const float* ff1_b   = (const float*)d_in[10];
  const float* ff2_w   = (const float*)d_in[11];
  const float* ff2_b   = (const float*)d_in[12];
  const float* ln2_g   = (const float*)d_in[13];
  const float* ln2_b   = (const float*)d_in[14];

  char* ws = (char*)d_ws;
  size_t off = 0;
  auto carve = [&](size_t bytes) -> char* {
    char* p = ws + off;
    off = (off + bytes + 255) & ~(size_t)255;
    return p;
  };

  _Float16* wqkv_h = (_Float16*)carve((size_t)L * D3 * D * sizeof(_Float16));
  _Float16* wo_h   = (_Float16*)carve((size_t)L * D  * D * sizeof(_Float16));
  _Float16* w1_h   = (_Float16*)carve((size_t)L * D4 * D * sizeof(_Float16));
  _Float16* w2_h   = (_Float16*)carve((size_t)L * D  * D4 * sizeof(_Float16));
  float*    x      = (float*)   carve((size_t)M * D  * sizeof(float));
  _Float16* xh     = (_Float16*)carve((size_t)M * D  * sizeof(_Float16));
  _Float16* qkvh   = (_Float16*)carve((size_t)M * D3 * sizeof(_Float16));
  _Float16* ah     = (_Float16*)carve((size_t)M * D  * sizeof(_Float16));
  _Float16* hh     = (_Float16*)carve((size_t)M * D4 * sizeof(_Float16));
  float*    yf     = (float*)   carve((size_t)M * D  * sizeof(float));

  // per-call weight conversion — deterministic; f16 weights are L2-resident
  f32_to_f16_kernel<<<4096, 256, 0, stream>>>(qkv_w, wqkv_h, (size_t)L * D3 * D);
  f32_to_f16_kernel<<<4096, 256, 0, stream>>>(out_w, wo_h,   (size_t)L * D  * D);
  f32_to_f16_kernel<<<4096, 256, 0, stream>>>(ff1_w, w1_h,   (size_t)L * D4 * D);
  f32_to_f16_kernel<<<4096, 256, 0, stream>>>(ff2_w, w2_h,   (size_t)L * D  * D4);

  embed_kernel<<<B, 256, 0, stream>>>(tokens, tok_emb, pos_emb, x, xh);

  for (int l = 0; l < L; ++l) {
    // QKV projection: [M,768] x [2304,768]^T -> f16 [M,2304]
    gemm_wmma_kernel<<<dim3(D3 / 256, M / 64), 256, 0, stream>>>(
        xh, wqkv_h + (size_t)l * D3 * D, qkv_b + (size_t)l * D3,
        nullptr, qkvh, D3, D, 0, 1);

    // flash attention -> f16 [M, 768]
    attn_kernel<<<dim3(S / 16, H, B), 32, 0, stream>>>(qkvh, tokens, ah);

    // out projection -> fp32 [M, 768]
    gemm_wmma_kernel<<<dim3(D / 256, M / 64), 256, 0, stream>>>(
        ah, wo_h + (size_t)l * D * D, out_b + (size_t)l * D,
        yf, nullptr, D, D, 0, 0);

    // x = LN(x + attn_proj)  (+ f16 mirror)
    add_ln_kernel<<<M, 256, 0, stream>>>(x, yf, ln1_g + (size_t)l * D,
                                         ln1_b + (size_t)l * D, x, xh);

    // FF1 + GELU -> f16 [M, 3072]
    gemm_wmma_kernel<<<dim3(D4 / 256, M / 64), 256, 0, stream>>>(
        xh, w1_h + (size_t)l * D4 * D, ff1_b + (size_t)l * D4,
        nullptr, hh, D4, D, 1, 1);

    // FF2 -> fp32 [M, 768]
    gemm_wmma_kernel<<<dim3(D / 256, M / 64), 256, 0, stream>>>(
        hh, w2_h + (size_t)l * D * D4, ff2_b + (size_t)l * D,
        yf, nullptr, D, D4, 0, 0);

    // x = LN(x + ff)  (+ f16 mirror)
    add_ln_kernel<<<M, 256, 0, stream>>>(x, yf, ln2_g + (size_t)l * D,
                                         ln2_b + (size_t)l * D, x, xh);
  }

  // outputs: x [B,S,D] fp32 then pad [B,S] as 0/1 floats
  float* out = (float*)d_out;
  copy_kernel<<<8192, 256, 0, stream>>>(x, out, (size_t)M * D);
  pad_out_kernel<<<(M + 255) / 256, 256, 0, stream>>>(tokens, out + (size_t)M * D, M);
}
